// TopkSparseAutoEncoder2Child_v2_37117107372098
// MI455X (gfx1250) — matrix operand order repaired and checked by reading, not compiled, not run on hardware
//
#include <hip/hip_runtime.h>
#include <hip/hip_bf16.h>

// ---------------------------------------------------------------------------
// TopkSparseAutoEncoder2Child for MI455X (gfx1250, wave32, WMMA + async LDS)
// B=2048, M=768, E=16384, K=32
// ---------------------------------------------------------------------------

#define BN 2048
#define MN 768
#define EN 16384
#define KN 32

typedef __bf16 bf16_t;
typedef __attribute__((ext_vector_type(16))) __bf16 v16bf;
typedef __attribute__((ext_vector_type(8)))  __bf16 v8bf;
typedef __attribute__((ext_vector_type(8)))  float  v8f;
typedef __attribute__((ext_vector_type(4)))  int    v4i;

// Detect gfx1250 async-LDS builtins (device pass only); fall back to a
// synchronous LDS copy if this toolchain doesn't declare them.
#if defined(__AMDGCN__)
#  if __has_builtin(__builtin_amdgcn_global_load_async_to_lds_b128) && \
      __has_builtin(__builtin_amdgcn_s_wait_asynccnt)
#    define USE_ASYNC_LDS 1
#  else
#    define USE_ASYNC_LDS 0
#  endif
#else
#  define USE_ASYNC_LDS 0
#endif

#if USE_ASYNC_LDS
// Builtin prototype (from hipcc diagnostic): (v4i AS1*, v4i AS3*, imm, imm)
#define ASYNC_B128(gp, lp)                                                  \
    __builtin_amdgcn_global_load_async_to_lds_b128(                         \
        (__attribute__((address_space(1))) v4i*)(gp),                       \
        (__attribute__((address_space(3))) v4i*)(lp), 0, 0)
#endif

// ---------------------------- workspace layout ------------------------------
// Region [0,144MB): first used as PRE (128MB, dead after top-k), then reused
// as the three transposed decoders (3 x 48MB). Stream ordering makes this safe.
static constexpr size_t OFF_PRE   = 0;
static constexpr size_t OFF_DECT0 = 0;
static constexpr size_t OFF_DECT1 = OFF_DECT0 + (size_t)EN * MN * 4;
static constexpr size_t OFF_DECT2 = OFF_DECT1 + (size_t)EN * MN * 4;
static constexpr size_t OFF_BASE  = (size_t)160 << 20;
static constexpr size_t OFF_XHI   = OFF_BASE;
static constexpr size_t OFF_XLO   = OFF_XHI + (size_t)BN * MN * 2;
static constexpr size_t OFF_WHI   = OFF_XLO + (size_t)BN * MN * 2;
static constexpr size_t OFF_WLO   = OFF_WHI + (size_t)EN * MN * 2;
static constexpr size_t OFF_VALS  = OFF_WLO + (size_t)EN * MN * 2;
static constexpr size_t OFF_IDX   = OFF_VALS + (size_t)BN * KN * 4;
static constexpr size_t OFF_F1    = OFF_IDX  + (size_t)BN * KN * 4;
static constexpr size_t OFF_F2    = OFF_F1   + (size_t)BN * KN * 4;
static constexpr size_t OFF_WIN   = OFF_F2   + (size_t)BN * KN * 4;
static constexpr size_t OFF_ST    = OFF_WIN  + (size_t)BN * KN * 4;  // 5*EN f32
static constexpr size_t OFF_FLAGS = OFF_ST   + (size_t)5 * EN * 4;   // 3*EN i32

static __device__ __forceinline__ v16bf cat16(v8bf a, v8bf b) {
    return __builtin_shufflevector(a, b, 0, 1, 2, 3, 4, 5, 6, 7,
                                         8, 9, 10, 11, 12, 13, 14, 15);
}

// ---------------------------------------------------------------------------
// 0) zero the live-flag arrays (workspace is NOT re-poisoned between replays)
__global__ void k_zero_i32(int* __restrict__ p, int n) {
    int i = blockIdx.x * blockDim.x + threadIdx.x;
    if (i < n) p[i] = 0;
}

// 1) fp32 -> bf16 hi/lo split (hi = rne(x), lo = rne(x - hi))
__global__ void k_split_bf16(const float* __restrict__ src,
                             bf16_t* __restrict__ hi, bf16_t* __restrict__ lo,
                             int n) {
    int i = blockIdx.x * blockDim.x + threadIdx.x;
    if (i < n) {
        float v  = src[i];
        bf16_t h = (bf16_t)v;
        hi[i] = h;
        lo[i] = (bf16_t)(v - (float)h);
    }
}

// ---------------------------------------------------------------------------
// 2) parent encoder GEMM: pre[b,e] = sum_m x[b,m]*enc_w[e,m] + enc_b[e]
//    Block: 256 threads (8 waves) -> 64 rows x 128 cols of output.
//    Wave: 64x16 (four 16x16 WMMA tiles sharing one B fragment per k-step).
//    A (hi/lo) staged into LDS with async-LDS loads, double-buffered in
//    64-wide K chunks; LDS rows padded to 72 bf16 to kill bank conflicts.
//    bf16 hi/lo split: 3 WMMAs per tile per k-step, fp32 accumulate.
#define CHUNK_K   64
#define NCHUNK    (MN / CHUNK_K)              // 12
#define LDS_STRIDE 72                         // 64 + 8 pad (bf16 elements)

__device__ __forceinline__ void stage_chunk(const bf16_t* __restrict__ srcHi,
                                            const bf16_t* __restrict__ srcLo,
                                            bf16_t* ldsHi, bf16_t* ldsLo,
                                            int rowBase, int k0, int tid) {
    // 64 rows x 64 k x 2B = 8KB per array = 512 16B units; 256 threads x 2.
#pragma unroll
    for (int i = 0; i < 2; ++i) {
        int u   = i * 256 + tid;
        int row = u >> 3;            // 8 units per row
        int c16 = u & 7;
        const bf16_t* gH = srcHi + (size_t)(rowBase + row) * MN + k0 + c16 * 8;
        const bf16_t* gL = srcLo + (size_t)(rowBase + row) * MN + k0 + c16 * 8;
        bf16_t* lH = ldsHi + row * LDS_STRIDE + c16 * 8;
        bf16_t* lL = ldsLo + row * LDS_STRIDE + c16 * 8;
#if USE_ASYNC_LDS
        ASYNC_B128(gH, lH);
        ASYNC_B128(gL, lL);
#else
        *(v8bf*)lH = *(const v8bf*)gH;
        *(v8bf*)lL = *(const v8bf*)gL;
#endif
    }
}

__global__ __launch_bounds__(256)
void k_enc_gemm(const bf16_t* __restrict__ xHi, const bf16_t* __restrict__ xLo,
                const bf16_t* __restrict__ wHi, const bf16_t* __restrict__ wLo,
                const float* __restrict__ bias, float* __restrict__ pre) {
    __shared__ bf16_t sA[2][2][64 * LDS_STRIDE];   // [buf][hi/lo] : 36,864 B

    const int tid   = threadIdx.x;
    const int lane  = tid & 31;
    const int wave  = tid >> 5;
    const int half  = lane >> 4;                   // 0: lanes 0-15, 1: 16-31
    const int l16   = lane & 15;
    const int rowBase = blockIdx.y * 64;
    const int eTile   = blockIdx.x * 8 + wave;
    const int e       = eTile * 16 + l16;

    // B (32x16 bf16): lane holds column n=l16 (== row e of enc_w);
    // K elements half*16 .. half*16+15 (contiguous).
    const bf16_t* bHbase = wHi + (size_t)e * MN + half * 16;
    const bf16_t* bLbase = wLo + (size_t)e * MN + half * 16;

    v8f acc[4] = {};

    stage_chunk(xHi, xLo, sA[0][0], sA[0][1], rowBase, 0, tid);

    for (int c = 0; c < NCHUNK; ++c) {
        const int cur = c & 1;
        if (c + 1 < NCHUNK)
            stage_chunk(xHi, xLo, sA[cur ^ 1][0], sA[cur ^ 1][1],
                        rowBase, (c + 1) * CHUNK_K, tid);
#if USE_ASYNC_LDS
        if (c + 1 < NCHUNK) __builtin_amdgcn_s_wait_asynccnt(4);
        else                __builtin_amdgcn_s_wait_asynccnt(0);
#endif
        __syncthreads();

        const bf16_t* aHi = sA[cur][0];
        const bf16_t* aLo = sA[cur][1];

#pragma unroll
        for (int kk = 0; kk < CHUNK_K; kk += 32) {
            const int kg = c * CHUNK_K + kk;
            __builtin_prefetch(bHbase + kg + 256, 0, 1);
            v8bf b0 = *(const v8bf*)(bHbase + kg);
            v8bf b1 = *(const v8bf*)(bHbase + kg + 8);
            v8bf q0 = *(const v8bf*)(bLbase + kg);
            v8bf q1 = *(const v8bf*)(bLbase + kg + 8);
            v16bf fBH = cat16(b0, b1);
            v16bf fBL = cat16(q0, q1);

#pragma unroll
            for (int r = 0; r < 4; ++r) {
                // A (16x32 bf16): lane = row (r*16+l16); elements
                // [kk + half*16 .. +15] (contiguous).
                const bf16_t* pH = aHi + (r * 16 + l16) * LDS_STRIDE + kk + half * 16;
                const bf16_t* pL = aLo + (r * 16 + l16) * LDS_STRIDE + kk + half * 16;
                v8bf a0 = *(const v8bf*)(pH);
                v8bf a1 = *(const v8bf*)(pH + 8);
                v8bf l0 = *(const v8bf*)(pL);
                v8bf l1 = *(const v8bf*)(pL + 8);
                v16bf fAH = cat16(a0, a1);
                v16bf fAL = cat16(l0, l1);
                acc[r] = __builtin_amdgcn_wmma_f32_16x16x32_bf16(
                             false, fAH, false, fBH, (short)0, acc[r], false, false);
                acc[r] = __builtin_amdgcn_wmma_f32_16x16x32_bf16(
                             false, fAH, false, fBL, (short)0, acc[r], false, false);
                acc[r] = __builtin_amdgcn_wmma_f32_16x16x32_bf16(
                             false, fAL, false, fBH, (short)0, acc[r], false, false);
            }
        }
        __syncthreads();
    }

    // D layout: VGPR q holds row (half*8 + q), column n = l16
    const float bb = bias[e];
#pragma unroll
    for (int r = 0; r < 4; ++r) {
#pragma unroll
        for (int q = 0; q < 8; ++q) {
            int row = rowBase + r * 16 + half * 8 + q;
            pre[(size_t)row * EN + e] = acc[r][q] + bb;
        }
    }
}

// ---------------------------------------------------------------------------
// 3) per-row top-32 (values + indices), descending.
//    128 threads/row: per-thread insertion top-32, then LDS merge tree.
__global__ __launch_bounds__(128)
void k_topk(const float* __restrict__ pre, float* __restrict__ vals,
            int* __restrict__ idx) {
    __shared__ float lv[128 * 32];
    __shared__ int   li[128 * 32];
    const int b = blockIdx.x;
    const int t = threadIdx.x;

    float v[32];
    int   id[32];
#pragma unroll
    for (int i = 0; i < 32; ++i) { v[i] = -__builtin_inff(); id[i] = -1; }

    const float* row = pre + (size_t)b * EN;
    for (int e = t; e < EN; e += 128) {
        float p = row[e];
        if (p > v[31]) {
            int pos = 31;
            while (pos > 0 && v[pos - 1] < p) {
                v[pos] = v[pos - 1]; id[pos] = id[pos - 1]; --pos;
            }
            v[pos] = p; id[pos] = e;
        }
    }
    for (int i = 0; i < 32; ++i) { lv[t * 32 + i] = v[i]; li[t * 32 + i] = id[i]; }
    __syncthreads();

    for (int stride = 64; stride >= 1; stride >>= 1) {
        if (t < stride) {
            float av[32], bv[32]; int ai[32], bi[32];
            for (int i = 0; i < 32; ++i) {
                av[i] = lv[t * 32 + i];            ai[i] = li[t * 32 + i];
                bv[i] = lv[(t + stride) * 32 + i]; bi[i] = li[(t + stride) * 32 + i];
            }
            float mv[32]; int mi[32];
            int pa = 0, pb = 0;
            for (int i = 0; i < 32; ++i) {   // pa+pb==i <= 31 at read time: in range
                if (av[pa] >= bv[pb]) { mv[i] = av[pa]; mi[i] = ai[pa]; ++pa; }
                else                  { mv[i] = bv[pb]; mi[i] = bi[pb]; ++pb; }
            }
            for (int i = 0; i < 32; ++i) { lv[t * 32 + i] = mv[i]; li[t * 32 + i] = mi[i]; }
        }
        __syncthreads();
    }
    if (t < 32) { vals[b * 32 + t] = lv[t]; idx[b * 32 + t] = li[t]; }
}

// ---------------------------------------------------------------------------
// 4) child pre-activations recomputed ONLY at the k active indices.
//    One wave per (b, j): 768-length dots + shuffle reduction.
__global__ __launch_bounds__(256)
void k_children(const float* __restrict__ x,
                const float* __restrict__ e1w, const float* __restrict__ e1b,
                const float* __restrict__ e2w, const float* __restrict__ e2b,
                const float* __restrict__ vals, const int* __restrict__ idx,
                float* __restrict__ f1, float* __restrict__ f2,
                int* __restrict__ win,
                int* __restrict__ flagsP, int* __restrict__ flags1,
                int* __restrict__ flags2) {
    const int wave = threadIdx.x >> 5, lane = threadIdx.x & 31;
    const int task = blockIdx.x * 8 + wave;        // 0 .. B*K-1
    const int b = task >> 5, j = task & 31;
    const int e = idx[b * 32 + j];
    const float a = vals[b * 32 + j];

    const float* xr = x   + (size_t)b * MN;
    const float* w1 = e1w + (size_t)e * MN;
    const float* w2 = e2w + (size_t)e * MN;
    float s1 = 0.f, s2 = 0.f;
    for (int m = lane; m < MN; m += 32) {
        float xv = xr[m];
        s1 += xv * w1[m];
        s2 += xv * w2[m];
    }
#pragma unroll
    for (int off = 16; off; off >>= 1) {
        s1 += __shfl_xor(s1, off, 32);
        s2 += __shfl_xor(s2, off, 32);
    }
    if (lane == 0) {
        float p1 = s1 + e1b[e], p2 = s2 + e2b[e];
        bool active = (a != 0.0f);
        float m1 = active ? p1 : 0.0f, m2 = active ? p2 : 0.0f;
        bool w = m1 > m2;
        float f1v = w ? m1 : 0.0f;
        float f2v = w ? 0.0f : m2;
        f1[b * 32 + j] = f1v;
        f2[b * 32 + j] = f2v;
        win[b * 32 + j] = w ? 1 : 0;
        atomicOr(&flagsP[e], 1);                 // parent counts every top-k index
        if (f1v != 0.0f) atomicOr(&flags1[e], 1);
        if (f2v != 0.0f) atomicOr(&flags2[e], 1);
    }
}

// ---------------------------------------------------------------------------
// 5) decoder transpose [M,E] -> [E,M] (coalesced reads)
__global__ void k_transpose(const float* __restrict__ src, float* __restrict__ dst) {
    size_t i = (size_t)blockIdx.x * blockDim.x + threadIdx.x;
    if (i < (size_t)MN * EN) {
        int m = (int)(i / EN), e = (int)(i % EN);
        dst[(size_t)e * MN + m] = src[i];
    }
}

// ---------------------------------------------------------------------------
// 6) per-column decoder stats: ||dp||^2, ||d1||^2, ||d2||^2, <dp,d1>, <dp,d2>
__global__ __launch_bounds__(256)
void k_colstats(const float* __restrict__ dT0, const float* __restrict__ dT1,
                const float* __restrict__ dT2,
                float* __restrict__ np2, float* __restrict__ n1,
                float* __restrict__ n2,  float* __restrict__ dp1,
                float* __restrict__ dp2) {
    const int wave = threadIdx.x >> 5, lane = threadIdx.x & 31;
    const int e = blockIdx.x * 8 + wave;
    const float* r0 = dT0 + (size_t)e * MN;
    const float* r1 = dT1 + (size_t)e * MN;
    const float* r2 = dT2 + (size_t)e * MN;
    float s0 = 0, s1 = 0, s2 = 0, s3 = 0, s4 = 0;
    for (int m = lane; m < MN; m += 32) {
        float a = r0[m], b1 = r1[m], b2 = r2[m];
        s0 += a * a; s1 += b1 * b1; s2 += b2 * b2; s3 += a * b1; s4 += a * b2;
    }
#pragma unroll
    for (int off = 16; off; off >>= 1) {
        s0 += __shfl_xor(s0, off, 32); s1 += __shfl_xor(s1, off, 32);
        s2 += __shfl_xor(s2, off, 32); s3 += __shfl_xor(s3, off, 32);
        s4 += __shfl_xor(s4, off, 32);
    }
    if (lane == 0) { np2[e] = s0; n1[e] = s1; n2[e] = s2; dp1[e] = s3; dp2[e] = s4; }
}

// ---------------------------------------------------------------------------
// 7) sparse decode: recon[b,:] = biases + sum_j a*dec[:,e] + f1*dec1[:,e] + f2*dec2[:,e]
__global__ __launch_bounds__(256)
void k_decode(const float* __restrict__ dT0, const float* __restrict__ dT1,
              const float* __restrict__ dT2,
              const float* __restrict__ db0, const float* __restrict__ db1,
              const float* __restrict__ db2,
              const float* __restrict__ vals, const int* __restrict__ idx,
              const float* __restrict__ f1, const float* __restrict__ f2,
              float* __restrict__ out) {
    const int b = blockIdx.x, t = threadIdx.x;
    __shared__ int   se[32];
    __shared__ float sa[32], s1[32], s2[32];
    if (t < 32) {
        se[t] = idx[b * 32 + t];
        sa[t] = vals[b * 32 + t];
        s1[t] = f1[b * 32 + t];
        s2[t] = f2[b * 32 + t];
    }
    __syncthreads();
    for (int m = t; m < MN; m += 256) {
        float acc = db0[m] + db1[m] + db2[m];
#pragma unroll 4
        for (int j = 0; j < 32; ++j) {
            size_t r = (size_t)se[j] * MN + m;
            acc += sa[j] * dT0[r] + s1[j] * dT1[r] + s2[j] * dT2[r];
        }
        out[(size_t)b * MN + m] = acc;
    }
}

// ---------------------------------------------------------------------------
// 8) finalize: live counts + aux loss -> d_out tail (single block)
__global__ __launch_bounds__(256)
void k_finalize(const int* __restrict__ flagsP, const int* __restrict__ flags1,
                const int* __restrict__ flags2,
                const float* __restrict__ vals, const int* __restrict__ idx,
                const float* __restrict__ f1, const float* __restrict__ f2,
                const int* __restrict__ win,
                const float* __restrict__ np2, const float* __restrict__ n1,
                const float* __restrict__ n2,  const float* __restrict__ dp1,
                const float* __restrict__ dp2,
                float* __restrict__ tail) {
    __shared__ float red[256];
    const int t = threadIdx.x;

    float cp = 0, c1 = 0, c2 = 0;
    for (int e = t; e < EN; e += 256) {
        cp += (float)flagsP[e];
        c1 += (float)flags1[e];
        c2 += (float)flags2[e];
    }
    float s = 0.f;
    for (int i = t; i < BN * KN; i += 256) {
        float a = vals[i];
        if (a > 0.0f) {
            int e = idx[i];
            int w = win[i];
            float c   = w ? f1[i]  : f2[i];
            float npv = np2[e];
            float dpc = w ? dp1[e] : dp2[e];
            float nc  = w ? n1[e]  : n2[e];
            float a2 = a * a;
            float dot   = a2 * npv + a * c * dpc;
            float normp = fabsf(a) * sqrtf(npv);
            float comb2 = a2 * npv + 2.0f * a * c * dpc + c * c * nc;
            float normc = sqrtf(fmaxf(comb2, 0.0f));
            s += dot / (fmaxf(normp, 1e-8f) * fmaxf(normc, 1e-8f));
        }
    }
    float parts[4] = { cp, c1, c2, -s / (float)BN };
#pragma unroll
    for (int slot = 0; slot < 4; ++slot) {
        red[t] = parts[slot];
        __syncthreads();
        for (int str = 128; str; str >>= 1) {
            if (t < str) red[t] += red[t + str];
            __syncthreads();
        }
        if (t == 0) tail[slot] = red[0];
        __syncthreads();
    }
}

// ---------------------------------------------------------------------------
extern "C" void kernel_launch(void* const* d_in, const int* in_sizes, int n_in,
                              void* d_out, int out_size, void* d_ws, size_t ws_size,
                              hipStream_t stream) {
    (void)in_sizes; (void)n_in; (void)out_size; (void)ws_size;
    const float* x      = (const float*)d_in[0];
    const float* enc_w  = (const float*)d_in[1];
    const float* enc_b  = (const float*)d_in[2];
    const float* dec_w  = (const float*)d_in[3];
    const float* dec_b  = (const float*)d_in[4];
    const float* enc1_w = (const float*)d_in[5];
    const float* enc1_b = (const float*)d_in[6];
    const float* dec1_w = (const float*)d_in[7];
    const float* dec1_b = (const float*)d_in[8];
    const float* enc2_w = (const float*)d_in[9];
    const float* enc2_b = (const float*)d_in[10];
    const float* dec2_w = (const float*)d_in[11];
    const float* dec2_b = (const float*)d_in[12];
    // d_in[13] == k (fixed 32)

    char* ws = (char*)d_ws;
    float*  pre   = (float*)(ws + OFF_PRE);
    float*  dT0   = (float*)(ws + OFF_DECT0);
    float*  dT1   = (float*)(ws + OFF_DECT1);
    float*  dT2   = (float*)(ws + OFF_DECT2);
    bf16_t* xHi   = (bf16_t*)(ws + OFF_XHI);
    bf16_t* xLo   = (bf16_t*)(ws + OFF_XLO);
    bf16_t* wHi   = (bf16_t*)(ws + OFF_WHI);
    bf16_t* wLo   = (bf16_t*)(ws + OFF_WLO);
    float*  vals  = (float*)(ws + OFF_VALS);
    int*    idx   = (int*)  (ws + OFF_IDX);
    float*  f1    = (float*)(ws + OFF_F1);
    float*  f2    = (float*)(ws + OFF_F2);
    int*    win   = (int*)  (ws + OFF_WIN);
    float*  np2   = (float*)(ws + OFF_ST);
    float*  n1    = np2 + EN;
    float*  n2    = np2 + 2 * EN;
    float*  dp1   = np2 + 3 * EN;
    float*  dp2   = np2 + 4 * EN;
    int*    flagsP = (int*)(ws + OFF_FLAGS);
    int*    flags1 = flagsP + EN;
    int*    flags2 = flagsP + 2 * EN;

    float* out   = (float*)d_out;             // recon [B, M]
    float* tail  = out + (size_t)BN * MN;     // counts + aux

    // 0) zero flags (ws is not re-poisoned between graph replays)
    k_zero_i32<<<(3 * EN + 255) / 256, 256, 0, stream>>>(flagsP, 3 * EN);

    // 1) bf16 hi/lo splits
    k_split_bf16<<<(BN * MN + 255) / 256, 256, 0, stream>>>(x, xHi, xLo, BN * MN);
    k_split_bf16<<<(EN * MN + 255) / 256, 256, 0, stream>>>(enc_w, wHi, wLo, EN * MN);

    // 2) parent encoder GEMM via WMMA: block = 64 rows x 128 cols
    k_enc_gemm<<<dim3(EN / 128, BN / 64), 256, 0, stream>>>(xHi, xLo, wHi, wLo,
                                                            enc_b, pre);

    // 3) top-32 per row
    k_topk<<<BN, 128, 0, stream>>>(pre, vals, idx);

    // 4) child values at active indices only (B*K waves)
    k_children<<<(BN * KN) / 8, 256, 0, stream>>>(x, enc1_w, enc1_b, enc2_w, enc2_b,
                                                  vals, idx, f1, f2, win,
                                                  flagsP, flags1, flags2);

    // 5) decoder transposes (reuse the dead PRE region)
    const int tgrid = (MN * EN + 255) / 256;
    k_transpose<<<tgrid, 256, 0, stream>>>(dec_w,  dT0);
    k_transpose<<<tgrid, 256, 0, stream>>>(dec1_w, dT1);
    k_transpose<<<tgrid, 256, 0, stream>>>(dec2_w, dT2);

    // 6) decoder column stats (E waves)
    k_colstats<<<EN / 8, 256, 0, stream>>>(dT0, dT1, dT2, np2, n1, n2, dp1, dp2);

    // 7) sparse decode -> recon
    k_decode<<<BN, 256, 0, stream>>>(dT0, dT1, dT2, dec_b, dec1_b, dec2_b,
                                     vals, idx, f1, f2, out);

    // 8) counts + aux loss
    k_finalize<<<1, 256, 0, stream>>>(flagsP, flags1, flags2, vals, idx, f1, f2,
                                      win, np2, n1, n2, dp1, dp2, tail);
}
